// MultiheadFlashDiff_26731876450795
// MI455X (gfx1250) — matrix operand order repaired and checked by reading, not compiled, not run on hardware
//
#include <hip/hip_runtime.h>
#include <hip/hip_bf16.h>
#include <math.h>

#define E_DIM       1024
#define N_HEADS     8
#define HEAD_D      64
#define SEQ         4096
#define ROWS        16384                    // b*s = 4*4096
#define LAMBDA_INIT 0.7836057665316245f      // 0.8 - 0.6*exp(-3.6)
#define LOG10000    9.210340371976184f

typedef __bf16 bf16_t;
typedef __attribute__((ext_vector_type(16))) __bf16 v16bf;
typedef __attribute__((ext_vector_type(8)))  __bf16 v8bf;
typedef __attribute__((ext_vector_type(4)))  __bf16 v4bf;
typedef __attribute__((ext_vector_type(8)))  float  v8f;
typedef __attribute__((ext_vector_type(4)))  float  v4f_t;

// ---------------------------------------------------------------- converts
__global__ void f32_to_bf16_kernel(const float* __restrict__ in,
                                   bf16_t* __restrict__ out, int n4) {
  int i = blockIdx.x * blockDim.x + threadIdx.x;
  if (i < n4) {
    v4f_t x = ((const v4f_t*)in)[i];
    v4bf y;
    y.x = (bf16_t)x.x; y.y = (bf16_t)x.y; y.z = (bf16_t)x.z; y.w = (bf16_t)x.w;
    ((v4bf*)out)[i] = y;
  }
}

// W (E x E row-major, fp32) -> WT (E x E row-major = W transposed, bf16)
__global__ void transpose_to_bf16_kernel(const float* __restrict__ W,
                                         bf16_t* __restrict__ WT) {
  int i = blockIdx.x * blockDim.x + threadIdx.x;   // 1M threads
  int n = i >> 10, k = i & 1023;
  WT[i] = (bf16_t)W[(size_t)k * E_DIM + n];
}

// ---------------------------------------------------------------- lambda
__global__ void lambda_kernel(const float* __restrict__ lq1, const float* __restrict__ lk1,
                              const float* __restrict__ lq2, const float* __restrict__ lk2,
                              float* __restrict__ lam_out) {
  __shared__ float s1[64], s2[64];
  int t = threadIdx.x;
  s1[t] = lq1[t] * lk1[t];
  s2[t] = lq2[t] * lk2[t];
  __syncthreads();
  if (t == 0) {
    float a = 0.f, b = 0.f;
    for (int i = 0; i < 64; ++i) { a += s1[i]; b += s2[i]; }
    *lam_out = __expf(a) - __expf(b) + LAMBDA_INIT;
  }
}

// ---------------------------------------------------------------- WMMA GEMM
// A-frag: lane l<16 = row l; elems 0..7 = K[k0+hi*8 .. +7], elems 8..15 = K[k0+16+hi*8 ..]
__device__ __forceinline__ v16bf load_a_frag(const bf16_t* __restrict__ base, int ld,
                                             int row0, int k0, int lane) {
  int l = lane & 15, hi = lane >> 4;
  const bf16_t* p = base + (size_t)(row0 + l) * ld + k0 + hi * 8;
  v8bf lo = *(const v8bf*)p;
  v8bf hh = *(const v8bf*)(p + 16);
  return __builtin_shufflevector(lo, hh, 0,1,2,3,4,5,6,7,8,9,10,11,12,13,14,15);
}

// B-frag (from transposed weight, row = output column): lane n<16 = col n,
// elems 0..7 = K[k0+hi*8 .. +7], elems 8..15 = K[k0+16+hi*8 ..]
__device__ __forceinline__ v16bf load_b_frag(const bf16_t* __restrict__ baseT, int ld,
                                             int col0, int k0, int lane) {
  int l = lane & 15, hi = lane >> 4;
  const bf16_t* p = baseT + (size_t)(col0 + l) * ld + k0 + hi * 8;
  v8bf lo = *(const v8bf*)p;
  v8bf hh = *(const v8bf*)(p + 16);
  return __builtin_shufflevector(lo, hh, 0,1,2,3,4,5,6,7,8,9,10,11,12,13,14,15);
}

template <bool STORE_BF16>
__device__ __forceinline__
void gemm_body(const bf16_t* __restrict__ A,   // M x K row-major
               const bf16_t* __restrict__ BT,  // N x K row-major (B transposed)
               void* __restrict__ C,           // M x N
               int N, int K, int m0, int n0, int lane) {
  v8f acc[4][4] = {};

  for (int k0 = 0; k0 < K; k0 += 32) {
    v16bf af[4], bfr[4];
#pragma unroll
    for (int i = 0; i < 4; ++i) af[i] = load_a_frag(A, K, m0 + i * 16, k0, lane);
#pragma unroll
    for (int j = 0; j < 4; ++j) bfr[j] = load_b_frag(BT, K, n0 + j * 16, k0, lane);

    if (k0 + 32 < K) {  // hint next K-slab into cache (global_prefetch_b8)
      __builtin_prefetch(A  + (size_t)(m0 + (lane & 15)) * K + k0 + 32, 0, 1);
      __builtin_prefetch(BT + (size_t)(n0 + (lane & 15)) * K + k0 + 32, 0, 1);
    }

#pragma unroll
    for (int i = 0; i < 4; ++i)
#pragma unroll
      for (int j = 0; j < 4; ++j)
        acc[i][j] = __builtin_amdgcn_wmma_f32_16x16x32_bf16(
            false, af[i], false, bfr[j], (short)0, acc[i][j], false, false);
  }

  const int l = lane & 15, hi = lane >> 4;
#pragma unroll
  for (int i = 0; i < 4; ++i) {
#pragma unroll
    for (int j = 0; j < 4; ++j) {
      int col  = n0 + j * 16 + l;
      int rowb = m0 + i * 16 + hi * 8;
#pragma unroll
      for (int r = 0; r < 8; ++r) {
        float vv = acc[i][j][r];
        if (STORE_BF16) ((bf16_t*)C)[(size_t)(rowb + r) * N + col] = (bf16_t)vv;
        else            ((float*) C)[(size_t)(rowb + r) * N + col] = vv;
      }
    }
  }
}

// One launch computes Q, K and V projections (blockIdx.z selects the weight).
__global__ __launch_bounds__(128)
void gemm_qkv_kernel(const bf16_t* __restrict__ xbf,
                     const bf16_t* __restrict__ WqT, const bf16_t* __restrict__ WkT,
                     const bf16_t* __restrict__ WvT,
                     bf16_t* __restrict__ qb, bf16_t* __restrict__ kb,
                     bf16_t* __restrict__ vb, int N, int K) {
  const bf16_t* BT = (blockIdx.z == 0) ? WqT : (blockIdx.z == 1) ? WkT : WvT;
  bf16_t*       Cc = (blockIdx.z == 0) ? qb  : (blockIdx.z == 1) ? kb  : vb;
  const int lane = threadIdx.x & 31;
  const int wave = threadIdx.x >> 5;
  const int m0 = blockIdx.y * 128 + (wave >> 1) * 64;
  const int n0 = blockIdx.x * 128 + (wave & 1) * 64;
  gemm_body<true>(xbf, BT, (void*)Cc, N, K, m0, n0, lane);
}

__global__ __launch_bounds__(128)
void gemm_out_kernel(const bf16_t* __restrict__ A, const bf16_t* __restrict__ BT,
                     float* __restrict__ C, int N, int K) {
  const int lane = threadIdx.x & 31;
  const int wave = threadIdx.x >> 5;
  const int m0 = blockIdx.y * 128 + (wave >> 1) * 64;
  const int n0 = blockIdx.x * 128 + (wave & 1) * 64;
  gemm_body<false>(A, BT, (void*)C, N, K, m0, n0, lane);
}

// ------------------------------------------- diff attn + fused rotary (per token)
__global__ __launch_bounds__(256)
void diff_attn_kernel(const bf16_t* __restrict__ q, const bf16_t* __restrict__ k,
                      const bf16_t* __restrict__ v, const float* __restrict__ lam_p,
                      const float* __restrict__ ln_w, const float* __restrict__ ln_b,
                      bf16_t* __restrict__ a_out) {
  __shared__ float qs[E_DIM], ks[E_DIM], vs[E_DIM], arow[E_DIM];
  __shared__ float sc[2][8][8];
  __shared__ float aw[2][8][8];
  __shared__ float rfac[8];

  const int tok = blockIdx.x;
  const int t   = threadIdx.x;
  const size_t rbase = (size_t)tok * E_DIM;

#pragma unroll
  for (int u = 0; u < 4; ++u) {
    int c = t + u * 256;
    qs[c] = (float)q[rbase + c];
    ks[c] = (float)k[rbase + c];
    vs[c] = (float)v[rbase + c];
  }
  __syncthreads();

  // Fused rotary on q,k in LDS: 512 interleaved pairs per row, subhead dim 64.
  const int pos = tok & (SEQ - 1);
#pragma unroll
  for (int u = 0; u < 2; ++u) {
    int p  = t + u * 256;          // pair index 0..511
    int c  = p * 2;
    int dd = c & 63;               // even dim index within 64-d subhead (== 2j)
    float invf  = __expf(-((float)dd / 64.0f) * LOG10000);
    float theta = (float)pos * invf;
    float sv, cv;
    __sincosf(theta, &sv, &cv);
    float x1 = qs[c], x2 = qs[c + 1];
    qs[c]     = x1 * cv - x2 * sv;
    qs[c + 1] = x2 * cv + x1 * sv;
    float y1 = ks[c], y2 = ks[c + 1];
    ks[c]     = y1 * cv - y2 * sv;
    ks[c + 1] = y2 * cv + y1 * sv;
  }
  __syncthreads();

  const float scale = 0.35355339059327373f;   // 64^-0.25
  if (t < 128) {                              // 2 sel * 8x8 scores, dot-64 each
    int sel = t >> 6, r = t & 63;
    int n = r >> 3, m = r & 7;
    int off = sel * 64;
    const float* qp = &qs[n * 128 + off];
    const float* kp = &ks[m * 128 + off];
    float s = 0.f;
#pragma unroll
    for (int d = 0; d < 64; ++d) s += qp[d] * kp[d];
    sc[sel][n][m] = s * scale;
  }
  __syncthreads();

  if (t < 16) {                               // softmax over m (8-wide)
    int sel = t >> 3, n = t & 7;
    float mx = sc[sel][n][0];
    for (int m = 1; m < 8; ++m) mx = fmaxf(mx, sc[sel][n][m]);
    float e[8], sum = 0.f;
    for (int m = 0; m < 8; ++m) { e[m] = __expf(sc[sel][n][m] - mx); sum += e[m]; }
    float inv = 1.0f / sum;
    for (int m = 0; m < 8; ++m) aw[sel][n][m] = e[m] * inv;
  }
  __syncthreads();

  const float lam = *lam_p;
#pragma unroll
  for (int u = 0; u < 4; ++u) {               // a = attn1 - lam*attn2
    int o = t + u * 256;
    int h = o >> 7, e = o & 127;
    float s1 = 0.f, s2 = 0.f;
#pragma unroll
    for (int m = 0; m < 8; ++m) {
      float vv = vs[m * 128 + e];
      s1 += aw[0][h][m] * vv;
      s2 += aw[1][h][m] * vv;
    }
    arow[o] = s1 - lam * s2;
  }
  __syncthreads();

  if (t < 8) {                                // RMS over 128-d per head
    float ss = 0.f;
    for (int e = 0; e < 128; ++e) { float x = arow[t * 128 + e]; ss += x * x; }
    rfac[t] = rsqrtf(ss * (1.0f / 128.0f) + 1e-8f);
  }
  __syncthreads();

  const float post = 1.0f - LAMBDA_INIT;
#pragma unroll
  for (int u = 0; u < 4; ++u) {
    int o = t + u * 256;
    int h = o >> 7, e = o & 127;
    float val = (arow[o] * rfac[h] * ln_w[e] + ln_b[e]) * post;
    a_out[rbase + o] = (bf16_t)val;
  }
}

// ---------------------------------------------------------------- launch
extern "C" void kernel_launch(void* const* d_in, const int* in_sizes, int n_in,
                              void* d_out, int out_size, void* d_ws, size_t ws_size,
                              hipStream_t stream) {
  const float* x    = (const float*)d_in[0];
  const float* Wq   = (const float*)d_in[1];
  const float* Wk   = (const float*)d_in[2];
  const float* Wv   = (const float*)d_in[3];
  const float* Wo   = (const float*)d_in[4];
  const float* lq1  = (const float*)d_in[5];
  const float* lk1  = (const float*)d_in[6];
  const float* lq2  = (const float*)d_in[7];
  const float* lk2  = (const float*)d_in[8];
  const float* ln_w = (const float*)d_in[9];
  const float* ln_b = (const float*)d_in[10];
  float* out = (float*)d_out;

  char* ws = (char*)d_ws;
  size_t off = 0;
  auto carve = [&](size_t bytes) -> void* {
    off = (off + 255) & ~(size_t)255;
    void* p = ws + off;
    off += bytes;
    return p;
  };
  const size_t XB = (size_t)ROWS * E_DIM * sizeof(bf16_t);   // 32 MB
  const size_t WB = (size_t)E_DIM * E_DIM * sizeof(bf16_t);  // 2 MB
  bf16_t* xbf  = (bf16_t*)carve(XB);
  bf16_t* WqT  = (bf16_t*)carve(WB);
  bf16_t* WkT  = (bf16_t*)carve(WB);
  bf16_t* WvT  = (bf16_t*)carve(WB);
  bf16_t* WoT  = (bf16_t*)carve(WB);
  bf16_t* qbf  = (bf16_t*)carve(XB);
  bf16_t* kbf  = (bf16_t*)carve(XB);
  bf16_t* vbf  = (bf16_t*)carve(XB);
  bf16_t* abf  = (bf16_t*)carve(XB);
  float*  lam  = (float*) carve(256);

  const int NX4 = ROWS * E_DIM / 4;                // 4,194,304 float4 groups
  f32_to_bf16_kernel<<<NX4 / 256, 256, 0, stream>>>(x, xbf, NX4);

  const int NW = E_DIM * E_DIM;
  transpose_to_bf16_kernel<<<NW / 256, 256, 0, stream>>>(Wq, WqT);
  transpose_to_bf16_kernel<<<NW / 256, 256, 0, stream>>>(Wk, WkT);
  transpose_to_bf16_kernel<<<NW / 256, 256, 0, stream>>>(Wv, WvT);
  transpose_to_bf16_kernel<<<NW / 256, 256, 0, stream>>>(Wo, WoT);

  lambda_kernel<<<1, 64, 0, stream>>>(lq1, lk1, lq2, lk2, lam);

  dim3 gqkv(E_DIM / 128, ROWS / 128, 3);           // (8, 128, 3)
  gemm_qkv_kernel<<<gqkv, 128, 0, stream>>>(xbf, WqT, WkT, WvT,
                                            qbf, kbf, vbf, E_DIM, E_DIM);

  diff_attn_kernel<<<ROWS, 256, 0, stream>>>(qbf, kbf, vbf, lam, ln_w, ln_b, abf);

  dim3 gout(E_DIM / 128, ROWS / 128);              // (8, 128)
  gemm_out_kernel<<<gout, 128, 0, stream>>>(abf, WoT, out, E_DIM, E_DIM);
}
